// FLARetNet_7756710936886
// MI455X (gfx1250) — matrix operand order repaired and checked by reading, not compiled, not run on hardware
//
#include <hip/hip_runtime.h>
#include <hip/hip_bf16.h>

// ---------------------------------------------------------------------------
// FLA RetNet forward for MI455X (gfx1250, wave32, WMMA bf16 16x16x32).
// B=4 T=2048 D=1024 H=16 DK=64 DV=128 CHUNK=64 N=32
// GEMMs: double-buffered async global->LDS (ASYNCcnt) feeding WMMA.
// ---------------------------------------------------------------------------

#define DEVFN __device__ __forceinline__

typedef __attribute__((ext_vector_type(16))) __bf16 v16bf;
typedef __attribute__((ext_vector_type(8)))  float  v8f;

constexpr int cB = 4, cT = 2048, cD = 1024, cH = 16;
constexpr int cDK = 64, cDV = 128, cCHUNK = 64, cN = 32;

// ---- bf16 helpers (round-to-nearest-even) ---------------------------------
DEVFN unsigned short f2b_raw(float f) {
  unsigned u = __builtin_bit_cast(unsigned, f);
  u += 0x7FFFu + ((u >> 16) & 1u);
  return (unsigned short)(u >> 16);
}
DEVFN __bf16 f2b(float f) {
  unsigned short s = f2b_raw(f);
  return __builtin_bit_cast(__bf16, s);
}
DEVFN float b2f(__bf16 b) {
  unsigned short s = __builtin_bit_cast(unsigned short, b);
  unsigned u = ((unsigned)s) << 16;
  return __builtin_bit_cast(float, u);
}

// ---- CDNA5 async global->LDS copy (ISA 10.x / CDNA5 ISA) ---------
// copies 16 B from gsrc to LDS offset l; tracked by ASYNCcnt.
DEVFN void async_b128(const __bf16* lds_dst, const __bf16* gsrc) {
  unsigned l = (unsigned)(uintptr_t)(const void*)lds_dst;  // low 32 = LDS off
  asm volatile("global_load_async_to_lds_b128 %0, %1, off"
               :: "v"(l), "v"(gsrc) : "memory");
}
DEVFN void wait_async0() {
  asm volatile("s_wait_asynccnt 0x0" ::: "memory");
}

// ---- WMMA fragment loads from LDS (row-major tiles) -----------------------
// A (16x32 bf16): lane L<16 -> row=L,  K = {0..7, 16..23}
//                 lane L>=16 -> row=L-16, K = {8..15, 24..31}
DEVFN v16bf load_a_frag(const __bf16* base, int stride, int row, int k0, int lane) {
  int r  = row + (lane & 15);
  int ko = (lane >> 4) << 3;           // 0 or 8
  union { v16bf v; uint4 u[2]; } f;
  f.u[0] = *(const uint4*)(base + (size_t)r * stride + k0 + ko);
  f.u[1] = *(const uint4*)(base + (size_t)r * stride + k0 + 16 + ko);
  return f.v;
}
// B (32x16 bf16), stored transposed in LDS as [col][K]:
// lane L<16 -> col=L, K = {0..7, 16..23} ; lane L>=16 -> col=L-16, K = {8..15, 24..31}
DEVFN v16bf load_b_frag(const __bf16* base, int stride, int col, int k0, int lane) {
  int c  = col + (lane & 15);
  int kb = (lane >> 4) << 3;           // 0 or 8
  union { v16bf v; uint4 u[2]; } f;
  f.u[0] = *(const uint4*)(base + (size_t)c * stride + k0 + kb);
  f.u[1] = *(const uint4*)(base + (size_t)c * stride + k0 + 16 + kb);
  return f.v;
}

DEVFN v8f wmma_bf16(v16bf a, v16bf b, v8f c) {
  return __builtin_amdgcn_wmma_f32_16x16x32_bf16(
      /*neg_a=*/false, a, /*neg_b=*/false, b,
      /*c_mod=*/(short)0, c, /*reuse_a=*/false, /*reuse_b=*/false);
}

// ---------------------------------------------------------------------------
// Kernel 0a: fp32 -> bf16 conversion (grid stride), for x.
// ---------------------------------------------------------------------------
__global__ void cvt_f32_bf16(const float* __restrict__ in,
                             __bf16* __restrict__ out, int n) {
  int i = blockIdx.x * blockDim.x + threadIdx.x;
  int stride = gridDim.x * blockDim.x;
  for (; i < n; i += stride) out[i] = f2b(in[i]);
}

// ---------------------------------------------------------------------------
// Kernel 0b: weight transpose+convert: in[K][N] f32 -> out[N][K] bf16.
// 32x32 LDS tile, block = 256 threads (32x8).
// ---------------------------------------------------------------------------
__global__ __launch_bounds__(256) void transpose_cvt(
    const float* __restrict__ in, __bf16* __restrict__ out, int K, int N) {
  __shared__ float t[32][33];
  const int n0 = blockIdx.x * 32, k0 = blockIdx.y * 32;
  const int tx = threadIdx.x & 31, ty = threadIdx.x >> 5;  // ty in 0..7
#pragma unroll
  for (int r = ty; r < 32; r += 8)
    t[r][tx] = in[(size_t)(k0 + r) * N + n0 + tx];
  __syncthreads();
#pragma unroll
  for (int r = ty; r < 32; r += 8)
    out[(size_t)(n0 + r) * K + k0 + tx] = f2b(t[tx][r]);
}

// ---------------------------------------------------------------------------
// Kernel 1: bf16 GEMM C[M,N] = A[M,K] @ W[K,N], W given TRANSPOSED (Wt[N][K]).
// Tile 128x128x64, double-buffered async global->LDS, 8 waves.
// Wave (wm=wid>>1, wn=wid&1) owns 2x4 16x16 accumulator tiles.
// mode: 0 = fp32 out, 1 = bf16 out, 2 = RoPE(q)+scale bf16, 3 = RoPE(k) bf16
// ---------------------------------------------------------------------------
enum { OUT_F32 = 0, OUT_BF16 = 1, OUT_ROPE_Q = 2, OUT_ROPE_K = 3 };

__global__ __launch_bounds__(256) void gemm_bf16(
    const __bf16* __restrict__ A, const __bf16* __restrict__ Wt,
    float* __restrict__ Cf, __bf16* __restrict__ Cb,
    int M, int N, int K, int mode, float scale) {
  extern __shared__ char gsm[];
  __bf16* As = (__bf16*)gsm;       // [2][128*64]  [row][k]
  __bf16* Bs = As + 2 * 128 * 64;  // [2][128*64]  [col][k]

  const int tid = threadIdx.x, lane = tid & 31, wid = tid >> 5;
  const int m0 = blockIdx.y * 128, n0 = blockIdx.x * 128;
  const int wm = wid >> 1, wn = wid & 1;

  v8f acc[2][4];
#pragma unroll
  for (int mi = 0; mi < 2; ++mi)
#pragma unroll
    for (int c = 0; c < 4; ++c) acc[mi][c] = (v8f){0, 0, 0, 0, 0, 0, 0, 0};

  // per-thread copy slots: 32 contiguous bf16 of A and of Wt per stage
  const int ar = tid >> 1, ak = (tid & 1) * 32;  // A: 128 rows x 64 k
  const int bc = tid >> 1, bk = (tid & 1) * 32;  // Wt: 128 cols x 64 k

  const int nt = K >> 6;  // K/64 stages

  // issue async copies for stage `st` into buffer `buf`
  auto issue = [&](int buf, int st) {
    const int kk = st << 6;
    const __bf16* asrc = A + (size_t)(m0 + ar) * K + kk + ak;
    __bf16* adst = As + buf * (128 * 64) + ar * 64 + ak;
    const __bf16* bsrc = Wt + (size_t)(n0 + bc) * K + kk + bk;
    __bf16* bdst = Bs + buf * (128 * 64) + bc * 64 + bk;
#pragma unroll
    for (int j = 0; j < 32; j += 8) {
      async_b128(adst + j, asrc + j);
      async_b128(bdst + j, bsrc + j);
    }
  };

  issue(0, 0);
  wait_async0();
  __syncthreads();

  for (int it = 0; it < nt; ++it) {
    const int cur = it & 1;
    if (it + 1 < nt) issue(cur ^ 1, it + 1);  // prefetch under the WMMAs

    const __bf16* Ab = As + cur * (128 * 64);
    const __bf16* Bb = Bs + cur * (128 * 64);
#pragma unroll
    for (int k0 = 0; k0 < 64; k0 += 32) {
      v16bf af[2], bfg[4];
#pragma unroll
      for (int mi = 0; mi < 2; ++mi)
        af[mi] = load_a_frag(Ab, 64, wm * 32 + mi * 16, k0, lane);
#pragma unroll
      for (int c = 0; c < 4; ++c)
        bfg[c] = load_b_frag(Bb, 64, wn * 64 + c * 16, k0, lane);
#pragma unroll
      for (int mi = 0; mi < 2; ++mi)
#pragma unroll
        for (int c = 0; c < 4; ++c)
          acc[mi][c] = wmma_bf16(af[mi], bfg[c], acc[mi][c]);
    }

    wait_async0();     // my async copies for next stage have landed
    __syncthreads();   // everyone done reading cur & writing nxt
  }

  // ---- epilogue --------------------------------------------------------
  const int hi8 = (lane >> 4) << 3;  // +8 rows for upper lane half
  if (mode == OUT_F32 || mode == OUT_BF16) {
#pragma unroll
    for (int mi = 0; mi < 2; ++mi)
#pragma unroll
      for (int c = 0; c < 4; ++c)
#pragma unroll
        for (int r = 0; r < 8; ++r) {
          int m = m0 + wm * 32 + mi * 16 + r + hi8;
          int n = n0 + wn * 64 + c * 16 + (lane & 15);
          float v = acc[mi][c][r];
          if (mode == OUT_F32) Cf[(size_t)m * N + n] = v;
          else                 Cb[(size_t)m * N + n] = f2b(v);
        }
  } else {
    // RoPE: head dim = 64, half = 32. Column n pairs with n+32, which is
    // accumulator tile c+2 of the SAME wave/lane/VGPR -> pure register rot.
#pragma unroll
    for (int mi = 0; mi < 2; ++mi)
#pragma unroll
      for (int c = 0; c < 2; ++c)
#pragma unroll
        for (int r = 0; r < 8; ++r) {
          int m = m0 + wm * 32 + mi * 16 + r + hi8;
          int n = n0 + wn * 64 + c * 16 + (lane & 15);
          int t = m & (cT - 1);                 // row = b*T + t
          int d = c * 16 + (lane & 15);         // 0..31
          float x1 = acc[mi][c][r];
          float x2 = acc[mi][c + 2][r];
          float invf = __expf(-(float)d * (0.2878231366242557f));  // ln(1e4)/32
          float ang = (float)t * invf;
          float s, co;
          __sincosf(ang, &s, &co);
          Cb[(size_t)m * N + n]      = f2b((x1 * co - x2 * s) * scale);
          Cb[(size_t)m * N + n + 32] = f2b((x2 * co + x1 * s) * scale);
        }
  }
}

// ---------------------------------------------------------------------------
// Kernel 2: chunked retention scan. One block per (b,h), 256 threads = 8 waves.
// S (64x128 fp32) persists in per-wave accumulators across 32 chunks.
// ---------------------------------------------------------------------------
__global__ __launch_bounds__(256) void retention(
    const __bf16* __restrict__ qg, const __bf16* __restrict__ kg,
    const __bf16* __restrict__ vg, float* __restrict__ o) {
  extern __shared__ char smem[];
  __bf16* q_s  = (__bf16*)smem;          // [64][64]  q
  __bf16* qc_s = q_s  + 64 * 64;         // [64][64]  q * cross[i]
  __bf16* kB_s = qc_s + 64 * 64;         // [64][64]  k  (B operand for q k^T)
  __bf16* kA_s = kB_s + 64 * 64;         // [64][64]  k^T * kdec (A operand)
  __bf16* a_s  = kA_s + 64 * 64;         // [64][64]  masked attn
  __bf16* vT_s = a_s  + 64 * 64;         // [128][64] v transposed
  __bf16* ST_s = vT_s + 128 * 64;        // [128][64] S^T bf16 (B operand)

  const int tid = threadIdx.x, lane = tid & 31, wid = tid >> 5;
  const int b = blockIdx.x / cH, h = blockIdx.x % cH;

  const float gamma = 1.0f - exp2f(-5.0f - (float)h);
  const float lg    = log2f(gamma);
  const float cdec  = exp2f(64.0f * lg);        // gamma^CHUNK

  v8f s_acc[4];
#pragma unroll
  for (int c = 0; c < 4; ++c) s_acc[c] = (v8f){0, 0, 0, 0, 0, 0, 0, 0};
  for (int i = tid; i < 128 * 64; i += 256) ST_s[i] = f2b(0.0f);
  __syncthreads();

  const int row_t = wid >> 1;          // 16-row band (rows / d-rows)
  const int colb  = (wid & 1) * 64;    // 64-col half of DV
  const int hi8   = (lane >> 4) << 3;

  for (int nch = 0; nch < cN; ++nch) {
    const int t0 = nch * cCHUNK;
    {  // ---- load chunk into LDS ----
      const int i = tid >> 2;
      const int d0 = (tid & 3) * 16;
      const float cross_i = exp2f(lg * (float)(i + 1));
      const float kdec_i  = exp2f(lg * (float)(63 - i));
      const size_t qoff =
          ((size_t)((b * cT + t0 + i) * cH + h)) * cDK + d0;
#pragma unroll
      for (int e = 0; e < 16; ++e) {
        __bf16 qv = qg[qoff + e];
        q_s[i * 64 + d0 + e]  = qv;
        qc_s[i * 64 + d0 + e] = f2b(b2f(qv) * cross_i);
        __bf16 kv = kg[qoff + e];
        kB_s[i * 64 + d0 + e]   = kv;
        kA_s[(d0 + e) * 64 + i] = f2b(b2f(kv) * kdec_i);
      }
      const int n0v = (tid & 3) * 32;
      const size_t voff =
          ((size_t)((b * cT + t0 + i) * cH + h)) * cDV + n0v;
#pragma unroll
      for (int e = 0; e < 32; ++e) vT_s[(n0v + e) * 64 + i] = vg[voff + e];
    }
    __syncthreads();

    {  // ---- attn = (q k^T) * Dmask -> a_s (bf16) ----
      const int ti  = wid >> 1;
      const int tj0 = (wid & 1) * 2;
#pragma unroll
      for (int tt = 0; tt < 2; ++tt) {
        const int tj = tj0 + tt;
        v8f acc = (v8f){0, 0, 0, 0, 0, 0, 0, 0};
#pragma unroll
        for (int k0 = 0; k0 < 64; k0 += 32)
          acc = wmma_bf16(load_a_frag(q_s, 64, ti * 16, k0, lane),
                          load_b_frag(kB_s, 64, tj * 16, k0, lane), acc);
#pragma unroll
        for (int r = 0; r < 8; ++r) {
          int i = ti * 16 + r + hi8;
          int j = tj * 16 + (lane & 15);
          float wgt = (i >= j) ? exp2f(lg * (float)(i - j)) : 0.0f;
          a_s[i * 64 + j] = f2b(acc[r] * wgt);
        }
      }
    }
    __syncthreads();

    {  // ---- o = attn @ v + (q*cross) @ S_old ----
#pragma unroll
      for (int c = 0; c < 4; ++c) {
        const int col = colb + c * 16;
        v8f acc = (v8f){0, 0, 0, 0, 0, 0, 0, 0};
#pragma unroll
        for (int k0 = 0; k0 < 64; k0 += 32)
          acc = wmma_bf16(load_a_frag(a_s, 64, row_t * 16, k0, lane),
                          load_b_frag(vT_s, 64, col, k0, lane), acc);
#pragma unroll
        for (int k0 = 0; k0 < 64; k0 += 32)
          acc = wmma_bf16(load_a_frag(qc_s, 64, row_t * 16, k0, lane),
                          load_b_frag(ST_s, 64, col, k0, lane), acc);
#pragma unroll
        for (int r = 0; r < 8; ++r) {
          int i  = row_t * 16 + r + hi8;
          int nn = col + (lane & 15);
          o[((size_t)((b * cT + t0 + i) * cH + h)) * cDV + nn] = acc[r];
        }
      }
    }
    __syncthreads();  // all waves done reading ST_s before overwrite

    {  // ---- S = S * gamma^CHUNK + (k*kdec)^T @ v ----
#pragma unroll
      for (int c = 0; c < 4; ++c) {
        const int col = colb + c * 16;
#pragma unroll
        for (int e = 0; e < 8; ++e) s_acc[c][e] *= cdec;
#pragma unroll
        for (int k0 = 0; k0 < 64; k0 += 32)
          s_acc[c] = wmma_bf16(load_a_frag(kA_s, 64, row_t * 16, k0, lane),
                               load_b_frag(vT_s, 64, col, k0, lane), s_acc[c]);
#pragma unroll
        for (int r = 0; r < 8; ++r) {
          int dd = row_t * 16 + r + hi8;
          int nn = col + (lane & 15);
          ST_s[nn * 64 + dd] = f2b(s_acc[c][r]);
        }
      }
    }
    __syncthreads();
  }
}

// ---------------------------------------------------------------------------
// Kernel 3: RMS norm over DV + SiLU gate. One wave per (b,t,h) row.
// ---------------------------------------------------------------------------
__global__ __launch_bounds__(256) void gate_norm(
    const float* __restrict__ o, const float* __restrict__ g,
    const float* __restrict__ gnw, __bf16* __restrict__ ob) {
  const int lane = threadIdx.x & 31, wid = threadIdx.x >> 5;
  const size_t row = (size_t)blockIdx.x * 8 + wid;
  const float4 ov = *(const float4*)(o + row * cDV + lane * 4);
  float ss = ov.x * ov.x + ov.y * ov.y + ov.z * ov.z + ov.w * ov.w;
#pragma unroll
  for (int off = 16; off > 0; off >>= 1) ss += __shfl_xor(ss, off, 32);
  const float rstd = rsqrtf(ss * (1.0f / 128.0f) + 1e-5f);
  const float4 gv = *(const float4*)(g + row * cDV + lane * 4);
  const float4 wv = *(const float4*)(gnw + lane * 4);
  const float ovals[4] = {ov.x, ov.y, ov.z, ov.w};
  const float gvals[4] = {gv.x, gv.y, gv.z, gv.w};
  const float wvals[4] = {wv.x, wv.y, wv.z, wv.w};
#pragma unroll
  for (int e = 0; e < 4; ++e) {
    float gg = gvals[e];
    float gate = gg / (1.0f + __expf(-gg));     // g * sigmoid(g)
    ob[row * cDV + lane * 4 + e] = f2b(ovals[e] * rstd * wvals[e] * gate);
  }
}

// ---------------------------------------------------------------------------
// Host launcher
// ---------------------------------------------------------------------------
extern "C" void kernel_launch(void* const* d_in, const int* in_sizes, int n_in,
                              void* d_out, int out_size, void* d_ws,
                              size_t ws_size, hipStream_t stream) {
  (void)in_sizes; (void)n_in; (void)out_size; (void)ws_size;
  const float* x   = (const float*)d_in[0];
  const float* Wq  = (const float*)d_in[1];
  const float* Wk  = (const float*)d_in[2];
  const float* Wv  = (const float*)d_in[3];
  const float* Wg  = (const float*)d_in[4];
  const float* Wo  = (const float*)d_in[5];
  const float* gnw = (const float*)d_in[6];

  char* ws = (char*)d_ws;
  __bf16* xb  = (__bf16*)(ws + 0);            // 16 MB  x bf16 [M][D]
  __bf16* Wqt = (__bf16*)(ws + 16777216);     //  2 MB  Wq^T [N][K]
  __bf16* Wkt = (__bf16*)(ws + 18874368);     //  2 MB
  __bf16* Wvt = (__bf16*)(ws + 20971520);     //  4 MB
  __bf16* Wgt = (__bf16*)(ws + 25165824);     //  4 MB
  __bf16* Wot = (__bf16*)(ws + 29360128);     //  4 MB
  __bf16* qb  = (__bf16*)(ws + 33554432);     // 16 MB  [B,T,H,DK]
  __bf16* kb  = (__bf16*)(ws + 50331648);     // 16 MB
  __bf16* vb  = (__bf16*)(ws + 67108864);     // 32 MB  [B,T,H,DV]
  float*  gf  = (float*) (ws + 100663296);    // 64 MB
  float*  of  = (float*) (ws + 167772160);    // 64 MB
  __bf16* obg = (__bf16*)(ws + 234881024);    // 32 MB  (end: 256 MB)

  const int M = cB * cT;  // 8192

  // x: fp32 -> bf16 ; weights: transpose + convert (for async b128 tile feed)
  cvt_f32_bf16<<<2048, 256, 0, stream>>>(x, xb, cB * cT * cD);
  transpose_cvt<<<dim3(32, 32), 256, 0, stream>>>(Wq, Wqt, cD, cH * cDK);
  transpose_cvt<<<dim3(32, 32), 256, 0, stream>>>(Wk, Wkt, cD, cH * cDK);
  transpose_cvt<<<dim3(64, 32), 256, 0, stream>>>(Wv, Wvt, cD, cH * cDV);
  transpose_cvt<<<dim3(64, 32), 256, 0, stream>>>(Wg, Wgt, cD, cH * cDV);
  transpose_cvt<<<dim3(32, 64), 256, 0, stream>>>(Wo, Wot, cH * cDV, cD);

  const size_t gemm_lds = 2u * 128 * 64 * 2 * 2;  // 64 KB double-buffered

  // projections
  gemm_bf16<<<dim3(8, M / 128), 256, gemm_lds, stream>>>(
      xb, Wqt, nullptr, qb, M, cH * cDK, cD, OUT_ROPE_Q, 0.125f);  // DK^-0.5
  gemm_bf16<<<dim3(8, M / 128), 256, gemm_lds, stream>>>(
      xb, Wkt, nullptr, kb, M, cH * cDK, cD, OUT_ROPE_K, 1.0f);
  gemm_bf16<<<dim3(16, M / 128), 256, gemm_lds, stream>>>(
      xb, Wvt, nullptr, vb, M, cH * cDV, cD, OUT_BF16, 1.0f);
  gemm_bf16<<<dim3(16, M / 128), 256, gemm_lds, stream>>>(
      xb, Wgt, gf, nullptr, M, cH * cDV, cD, OUT_F32, 1.0f);

  // chunked retention scan: one block per (b,h); 73728 B dynamic LDS
  retention<<<cB * cH, 256, 73728, stream>>>(qb, kb, vb, of);

  // norm + gate
  gate_norm<<<(cB * cT * cH) / 8, 256, 0, stream>>>(of, gf, gnw, obg);

  // output projection -> fp32 d_out
  gemm_bf16<<<dim3(8, M / 128), 256, gemm_lds, stream>>>(
      obg, Wot, (float*)d_out, nullptr, M, cD, cH * cDV, OUT_F32, 1.0f);
}
